// MonoMoE_77386720739415
// MI455X (gfx1250) — matrix operand order repaired and checked by reading, not compiled, not run on hardware
//
#include <hip/hip_runtime.h>
#include <hip/hip_bf16.h>

// ---------------------------------------------------------------------------
// MoE-VAE forward, fused, bf16 WMMA (gfx1250 / CDNA5, wave32)
// 512 threads = 16 waves; each wave owns a 32-row x 64-col output block
// (2x4 WMMA tiles, 64 acc VGPRs). K-loops unrolled 2x only (no spills).
// x tile staged to LDS as bf16 once per workgroup.
// ---------------------------------------------------------------------------

typedef __attribute__((ext_vector_type(16))) __bf16 v16bf;
typedef __attribute__((ext_vector_type(8)))  __bf16 v8bf;
typedef __attribute__((ext_vector_type(4)))  __bf16 v4bf;
typedef __attribute__((ext_vector_type(8)))  float  v8f;

#define E_ 8
#define B_ 16384
#define D_ 512
#define H_ 512
#define Z_ 64

// bf16 weight workspace layout (elements, per expert), all stored transposed [N][K]
constexpr size_t OFF_ENC_IN  = 0;                    // [512][512]
constexpr size_t OFF_ENC_H0  = 262144;               // [512][512]
constexpr size_t OFF_ENC_H1  = 524288;               // [512][512]
constexpr size_t OFF_MULV    = 786432;               // [128][512]  rows 0-63 mu^T, 64-127 lv^T
constexpr size_t OFF_DEC_IN  = 851968;               // [512][64]
constexpr size_t OFF_DEC_H0  = 884736;               // [512][512]
constexpr size_t OFF_DEC_H1  = 1146880;              // [512][512]
constexpr size_t OFF_DEC_OUT = 1409024;              // [512][512]
constexpr size_t WS_PER_E    = 1671168;

constexpr int HSTR  = 520;   // bf16 row stride for 64x512 activation tiles (bank-conflict pad)
constexpr int MVSTR = 132;   // f32 row stride for mu|lv tile
constexpr int ZSTR  = 72;    // bf16 row stride for z tile

__device__ __forceinline__ v8f wmma_bf16(v16bf a, v16bf b, v8f c) {
  return __builtin_amdgcn_wmma_f32_16x16x32_bf16(false, a, false, b, (short)0, c,
                                                 false, false);
}

// GEMM: acc[2][NPW] += A[rows mrow0..mrow0+31 of 64xK bf16 LDS tile] * Bt ([N][K] bf16)
// wave covers output columns [nbase, nbase + NPW*16)
template<int K, int NPW>
__device__ __forceinline__ void gemm_lds(v8f acc[2][NPW], const __bf16* A, int astr,
                                         int mrow0, const __bf16* Bt, int nbase,
                                         int l16, int hi) {
  const __bf16* bp[NPW];
#pragma unroll
  for (int ni = 0; ni < NPW; ni++)
    bp[ni] = Bt + (size_t)(nbase + ni * 16 + l16) * K + hi * 16;
  const __bf16* a0 = A + (mrow0 + l16) * astr + hi * 16;       // mi = 0
  const __bf16* a1 = a0 + 16 * astr;                            // mi = 1
#pragma unroll 2
  for (int k0 = 0; k0 < K; k0 += 32) {
    v16bf b[NPW];
#pragma unroll
    for (int ni = 0; ni < NPW; ni++)
      b[ni] = *(const v16bf*)(bp[ni] + k0);
    union { v16bf v; v8bf h2[2]; } u0, u1;
    u0.h2[0] = *(const v8bf*)(a0 + k0);
    u0.h2[1] = *(const v8bf*)(a0 + k0 + 8);
    u1.h2[0] = *(const v8bf*)(a1 + k0);
    u1.h2[1] = *(const v8bf*)(a1 + k0 + 8);
#pragma unroll
    for (int ni = 0; ni < NPW; ni++)
      acc[0][ni] = wmma_bf16(u0.v, b[ni], acc[0][ni]);
#pragma unroll
    for (int ni = 0; ni < NPW; ni++)
      acc[1][ni] = wmma_bf16(u1.v, b[ni], acc[1][ni]);
  }
}

// bias + relu + store bf16 tile to LDS (next layer's A)
__device__ __forceinline__ void relu_store(const v8f acc[2][4], const float* bias,
                                           int mrow0, int nbase, __bf16* out,
                                           int l16, int hi) {
#pragma unroll
  for (int ni = 0; ni < 4; ni++) {
    float bv = bias[nbase + ni * 16 + l16];
#pragma unroll
    for (int mi = 0; mi < 2; mi++)
#pragma unroll
      for (int j = 0; j < 8; j++) {
        float v = acc[mi][ni][j] + bv;
        v = v > 0.f ? v : 0.f;
        out[(mrow0 + mi * 16 + hi * 8 + j) * HSTR + nbase + ni * 16 + l16] = (__bf16)v;
      }
  }
}

__global__ __launch_bounds__(512) void moe_vae_kernel(
    const float* __restrict__ x, const float* __restrict__ eps,
    const __bf16* __restrict__ ws,
    const float* __restrict__ enc_in_b, const float* __restrict__ enc_hid_b,
    const float* __restrict__ mu_b, const float* __restrict__ lv_b,
    const float* __restrict__ dec_in_b, const float* __restrict__ dec_hid_b,
    const float* __restrict__ dec_out_b,
    float* __restrict__ out_mu, float* __restrict__ out_lv,
    float* __restrict__ out_xh) {
  extern __shared__ char smem[];
  __bf16* xs   = (__bf16*)smem;                    // 64*520 bf16 (x tile, persistent)
  __bf16* hA   = xs + 64 * HSTR;                   // 64*520 bf16
  __bf16* hB   = hA + 64 * HSTR;                   // 64*520 bf16
  float*  mulv = (float*)(hB + 64 * HSTR);         // 64*132 f32
  __bf16* zs   = (__bf16*)(mulv + 64 * MVSTR);     // 64*72 bf16
  float*  recon = (float*)(zs + 64 * ZSTR);        // 64 f32
  float*  bestr = recon + 64;                      // 64 f32
  int*    improved = (int*)(bestr + 64);           // 64 int

  const int tid  = threadIdx.x;
  const int wave = tid >> 5;
  const int lane = tid & 31;
  const int l16  = lane & 15;
  const int hi   = lane >> 4;
  const int g0   = blockIdx.x * 64;      // batch-row base for this tile
  const int mg   = wave >> 3;            // 0..1 : row-half of the 64-row tile
  const int ng   = wave & 7;             // 0..7 : 64-col block
  const int mrow0 = mg * 32;
  const int nb64  = ng * 64;

  if (tid < 64) { bestr[tid] = 3.4e38f; recon[tid] = 0.f; }
  // stage x tile to LDS as bf16 once (vectorized: 4 elems/thread/iter)
  for (int idx = tid; idx < 64 * 128; idx += 512) {
    const int m = idx >> 7, c4 = (idx & 127) * 4;
    const float4 v = *(const float4*)(x + (size_t)(g0 + m) * D_ + c4);
    v4bf b;
    b[0] = (__bf16)v.x; b[1] = (__bf16)v.y; b[2] = (__bf16)v.z; b[3] = (__bf16)v.w;
    *(v4bf*)(xs + m * HSTR + c4) = b;
  }
  __syncthreads();

  for (int e = 0; e < E_; ++e) {
    const __bf16* we = ws + (size_t)e * WS_PER_E;

    { // L1: h = relu(x @ enc_in_w + b)
      v8f acc[2][4] = {};
      gemm_lds<512, 4>(acc, xs, HSTR, mrow0, we + OFF_ENC_IN, nb64, l16, hi);
      relu_store(acc, enc_in_b + e * H_, mrow0, nb64, hA, l16, hi);
    }
    __syncthreads();
    { // L2: enc hidden 0
      v8f acc[2][4] = {};
      gemm_lds<512, 4>(acc, hA, HSTR, mrow0, we + OFF_ENC_H0, nb64, l16, hi);
      relu_store(acc, enc_hid_b + (size_t)(e * 2 + 0) * H_, mrow0, nb64, hB, l16, hi);
    }
    __syncthreads();
    { // L3: enc hidden 1
      v8f acc[2][4] = {};
      gemm_lds<512, 4>(acc, hB, HSTR, mrow0, we + OFF_ENC_H1, nb64, l16, hi);
      relu_store(acc, enc_hid_b + (size_t)(e * 2 + 1) * H_, mrow0, nb64, hA, l16, hi);
    }
    __syncthreads();
    { // L4: [mu | lv] = h @ [mu_w | lv_w]  (128 cols -> 4M x 8N tiles over 16 waves)
      v8f acc[2][1] = {};
      const int nb16 = ng * 16;
      gemm_lds<512, 1>(acc, hA, HSTR, mrow0, we + OFF_MULV, nb16, l16, hi);
      const int c = nb16 + l16;
      const float bv = (c < 64) ? mu_b[e * Z_ + c] : lv_b[e * Z_ + (c - 64)];
#pragma unroll
      for (int mi = 0; mi < 2; mi++)
#pragma unroll
        for (int j = 0; j < 8; j++)
          mulv[(mrow0 + mi * 16 + hi * 8 + j) * MVSTR + c] = acc[mi][0][j] + bv;
    }
    __syncthreads();
    // reparameterize: z = mu + exp(0.5*lv)*eps
    for (int idx = tid; idx < 64 * Z_; idx += 512) {
      const int m = idx >> 6, c = idx & 63;
      const float muv = mulv[m * MVSTR + c];
      const float lvv = mulv[m * MVSTR + 64 + c];
      const float zz  = muv + __expf(0.5f * lvv) * eps[(size_t)(g0 + m) * Z_ + c];
      zs[m * ZSTR + c] = (__bf16)zz;
    }
    __syncthreads();
    { // L5: g = relu(z @ dec_in_w + b), K=64
      v8f acc[2][4] = {};
      gemm_lds<64, 4>(acc, zs, ZSTR, mrow0, we + OFF_DEC_IN, nb64, l16, hi);
      relu_store(acc, dec_in_b + e * H_, mrow0, nb64, hA, l16, hi);
    }
    __syncthreads();
    { // L6: dec hidden 0
      v8f acc[2][4] = {};
      gemm_lds<512, 4>(acc, hA, HSTR, mrow0, we + OFF_DEC_H0, nb64, l16, hi);
      relu_store(acc, dec_hid_b + (size_t)(e * 2 + 0) * H_, mrow0, nb64, hB, l16, hi);
    }
    __syncthreads();
    { // L7: dec hidden 1
      v8f acc[2][4] = {};
      gemm_lds<512, 4>(acc, hB, HSTR, mrow0, we + OFF_DEC_H1, nb64, l16, hi);
      relu_store(acc, dec_hid_b + (size_t)(e * 2 + 1) * H_, mrow0, nb64, hA, l16, hi);
    }
    __syncthreads();
    { // L8: xhat = g @ dec_out_w + b; recon MSE; best-expert conditional writeout
      v8f acc[2][4] = {};
      gemm_lds<512, 4>(acc, hA, HSTR, mrow0, we + OFF_DEC_OUT, nb64, l16, hi);
      float bv[4];
#pragma unroll
      for (int ni = 0; ni < 4; ni++) bv[ni] = dec_out_b[e * D_ + nb64 + ni * 16 + l16];
      // per-row squared-error partials: reduce over 16 lanes, then across waves in LDS
#pragma unroll
      for (int mi = 0; mi < 2; mi++) {
#pragma unroll
        for (int j = 0; j < 8; j++) {
          const int m = mrow0 + mi * 16 + hi * 8 + j;
          const float* xr = x + (size_t)(g0 + m) * D_ + nb64 + l16;
          float s = 0.f;
#pragma unroll
          for (int ni = 0; ni < 4; ni++) {
            const float xh = acc[mi][ni][j] + bv[ni];
            const float d  = xh - xr[ni * 16];
            s += d * d;
          }
          s += __shfl_xor(s, 1, 16);
          s += __shfl_xor(s, 2, 16);
          s += __shfl_xor(s, 4, 16);
          s += __shfl_xor(s, 8, 16);
          if (l16 == 0) atomicAdd(&recon[m], s);
        }
      }
      __syncthreads();
      if (tid < 64) {
        const float r = recon[tid];
        const int imp = (r < bestr[tid]) ? 1 : 0;   // strict < == first-min argmin
        improved[tid] = imp;
        if (imp) bestr[tid] = r;
        recon[tid] = 0.f;
      }
      __syncthreads();
      // xhat straight from accumulators for improved rows
#pragma unroll
      for (int mi = 0; mi < 2; mi++) {
#pragma unroll
        for (int j = 0; j < 8; j++) {
          const int m = mrow0 + mi * 16 + hi * 8 + j;
          if (improved[m]) {
            float* orow = out_xh + (size_t)(g0 + m) * D_ + nb64 + l16;
#pragma unroll
            for (int ni = 0; ni < 4; ni++)
              orow[ni * 16] = acc[mi][ni][j] + bv[ni];
          }
        }
      }
      // mu / logvar from LDS for improved rows
      for (int idx = tid; idx < 64 * 128; idx += 512) {
        const int m = idx >> 7, c = idx & 127;
        if (improved[m]) {
          const float v = mulv[m * MVSTR + c];
          if (c < 64) out_mu[(size_t)(g0 + m) * Z_ + c] = v;
          else        out_lv[(size_t)(g0 + m) * Z_ + (c - 64)] = v;
        }
      }
    }
    __syncthreads();
  }
}

// ---------------------------------------------------------------------------
// Prep: fp32 [K][N] -> bf16 [N][K] (transpose + convert), blockIdx.z = expert
// ---------------------------------------------------------------------------
__global__ __launch_bounds__(256) void transpose_cvt(const float* __restrict__ src,
                                                     __bf16* __restrict__ dst,
                                                     int K, int N,
                                                     size_t src_stride_e,
                                                     size_t dst_stride_e) {
  __shared__ float tile[32][33];
  src += blockIdx.z * src_stride_e;
  dst += blockIdx.z * dst_stride_e;
  const int k0 = blockIdx.y * 32, n0 = blockIdx.x * 32;
  const int tx = threadIdx.x & 31, ty = threadIdx.x >> 5;  // 32 x 8
#pragma unroll
  for (int r = 0; r < 32; r += 8)
    tile[ty + r][tx] = src[(size_t)(k0 + ty + r) * N + n0 + tx];
  __syncthreads();
#pragma unroll
  for (int r = 0; r < 32; r += 8)
    dst[(size_t)(n0 + ty + r) * K + k0 + tx] = (__bf16)tile[tx][ty + r];
}

// ---------------------------------------------------------------------------
extern "C" void kernel_launch(void* const* d_in, const int* in_sizes, int n_in,
                              void* d_out, int out_size, void* d_ws, size_t ws_size,
                              hipStream_t stream) {
  (void)in_sizes; (void)n_in; (void)out_size; (void)ws_size;
  const float* x         = (const float*)d_in[0];
  const float* eps       = (const float*)d_in[1];
  const float* enc_in_w  = (const float*)d_in[2];
  const float* enc_in_b  = (const float*)d_in[3];
  const float* enc_hid_w = (const float*)d_in[4];
  const float* enc_hid_b = (const float*)d_in[5];
  const float* mu_w      = (const float*)d_in[6];
  const float* mu_b      = (const float*)d_in[7];
  const float* lv_w      = (const float*)d_in[8];
  const float* lv_b      = (const float*)d_in[9];
  const float* dec_in_w  = (const float*)d_in[10];
  const float* dec_in_b  = (const float*)d_in[11];
  const float* dec_hid_w = (const float*)d_in[12];
  const float* dec_hid_b = (const float*)d_in[13];
  const float* dec_out_w = (const float*)d_in[14];
  const float* dec_out_b = (const float*)d_in[15];

  __bf16* ws = (__bf16*)d_ws;     // needs 8 * 1671168 * 2 B = 25.5 MB
  float* out_mu = (float*)d_out;
  float* out_lv = out_mu + (size_t)B_ * Z_;
  float* out_xh = out_lv + (size_t)B_ * Z_;

  const dim3 blk(256);
  // one-time weight convert+transpose into L2-resident bf16 workspace
  transpose_cvt<<<dim3(16, 16, E_), blk, 0, stream>>>(enc_in_w, ws + OFF_ENC_IN, 512, 512, (size_t)512 * 512, WS_PER_E);
  transpose_cvt<<<dim3(16, 16, E_), blk, 0, stream>>>(enc_hid_w, ws + OFF_ENC_H0, 512, 512, (size_t)2 * 512 * 512, WS_PER_E);
  transpose_cvt<<<dim3(16, 16, E_), blk, 0, stream>>>(enc_hid_w + 512 * 512, ws + OFF_ENC_H1, 512, 512, (size_t)2 * 512 * 512, WS_PER_E);
  transpose_cvt<<<dim3(2, 16, E_),  blk, 0, stream>>>(mu_w, ws + OFF_MULV, 512, 64, (size_t)512 * 64, WS_PER_E);
  transpose_cvt<<<dim3(2, 16, E_),  blk, 0, stream>>>(lv_w, ws + OFF_MULV + (size_t)64 * 512, 512, 64, (size_t)512 * 64, WS_PER_E);
  transpose_cvt<<<dim3(16, 2, E_),  blk, 0, stream>>>(dec_in_w, ws + OFF_DEC_IN, 64, 512, (size_t)64 * 512, WS_PER_E);
  transpose_cvt<<<dim3(16, 16, E_), blk, 0, stream>>>(dec_hid_w, ws + OFF_DEC_H0, 512, 512, (size_t)2 * 512 * 512, WS_PER_E);
  transpose_cvt<<<dim3(16, 16, E_), blk, 0, stream>>>(dec_hid_w + 512 * 512, ws + OFF_DEC_H1, 512, 512, (size_t)2 * 512 * 512, WS_PER_E);
  transpose_cvt<<<dim3(16, 16, E_), blk, 0, stream>>>(dec_out_w, ws + OFF_DEC_OUT, 512, 512, (size_t)512 * 512, WS_PER_E);

  const size_t smem = (size_t)64 * HSTR * 2 * 3   // xs + hA + hB (bf16)
                    + (size_t)64 * MVSTR * 4      // mu|lv (f32)
                    + (size_t)64 * ZSTR * 2       // z (bf16)
                    + 3 * 64 * 4;                 // recon / best / improved
  hipFuncSetAttribute(reinterpret_cast<const void*>(moe_vae_kernel),
                      hipFuncAttributeMaxDynamicSharedMemorySize, (int)smem);

  moe_vae_kernel<<<dim3(B_ / 64), dim3(512), smem, stream>>>(
      x, eps, ws, enc_in_b, enc_hid_b, mu_b, lv_b, dec_in_b, dec_hid_b,
      dec_out_b, out_mu, out_lv, out_xh);
}
